// EntityResolutionProcessor_26972394619661
// MI455X (gfx1250) — matrix operand order repaired and checked by reading, not compiled, not run on hardware
//
#include <hip/hip_runtime.h>
#include <hip/hip_bf16.h>
#include <cstdint>

#define D_ 768
#define S_ 4096
#define M_ 1024
#define KC 32
#define MK 32768       /* M_*KC */
#define ROWS2 65536    /* MK*2  */
#define FFN_H 3072
#define CHUNK_ROWS 8192
#define NCHUNK 8

typedef __attribute__((ext_vector_type(16))) _Float16 v16h;
typedef __attribute__((ext_vector_type(8)))  float    v8f;
typedef __attribute__((ext_vector_type(4)))  unsigned int u32x4;

union FragU { v16h v; u32x4 q[2]; };
union Pack8 { u32x4 q; _Float16 h[8]; };

// ---------------------------------------------------------------------------
// Generic 128x128-tile WMMA GEMM: C = A(rows x Kd, f16) @ B(Kd x N, f16)
// MODE: 0 = store f32 (no bias) | 1 = store f16 (+bias, opt RELU)
//       2 = fused epilogue: relu(acc+bias+P[row>>5]) dot vecW -> atomicAdd accOut[row]
//       3 = acc+bias+token-residual (mention/cand) -> f32
//       4 = acc+bias+resFull -> f32
// A tile is copied global->LDS with GLOBAL_LOAD_ASYNC_TO_LDS_B128 (ASYNCcnt);
// B tile is register-transposed into k-major LDS for 2x ds_load_b128 frags.
// ---------------------------------------------------------------------------
template <int MODE, int RELU>
__global__ __launch_bounds__(256) void gemm_wmma_f16(
    const _Float16* __restrict__ A, const _Float16* __restrict__ B,
    int rows, int N, int Kd,
    const float* __restrict__ bias, const float* __restrict__ P,
    const float* __restrict__ vecW, float* __restrict__ accOut,
    float* __restrict__ outF, _Float16* __restrict__ outH,
    const float* __restrict__ resMention, const float* __restrict__ resCand,
    const float* __restrict__ resFull)
{
    __shared__ __align__(16) _Float16 As[128 * 40];
    __shared__ __align__(16) _Float16 Bs[128 * 40];

    const int tid  = threadIdx.x;
    const int lane = tid & 31;
    const int wave = tid >> 5;
    const int waveRow = wave & 3;   // 4 strips of 32 rows
    const int waveCol = wave >> 2;  // 2 strips of 64 cols
    const int l15   = lane & 15;
    const int lHalf = lane >> 4;

    const int rowBase = blockIdx.y * 128;
    const int colBase = blockIdx.x * 128;

    v8f zero = {0.f,0.f,0.f,0.f,0.f,0.f,0.f,0.f};
    v8f acc[2][4];
    #pragma unroll
    for (int i = 0; i < 2; ++i)
        #pragma unroll
        for (int j = 0; j < 4; ++j) acc[i][j] = zero;

    const int aRow = tid >> 1, aSeg = tid & 1;   // A: 128 rows x 32 k, 32 B/thread
    const int bK   = tid >> 3, bSeg = tid & 7;   // B: 32 k x 128 cols, 16 f16/thread

    // LDS byte address for this thread's A destination (low 32 bits of flat ptr)
    const unsigned aLds = (unsigned)(uintptr_t)(&As[aRow * 40 + aSeg * 16]);

    for (int ko = 0; ko < Kd; ko += 32) {
        const _Float16* ag = A + (size_t)(rowBase + aRow) * Kd + ko + aSeg * 16;
        const _Float16* bg = B + (size_t)(ko + bK) * N + colBase + bSeg * 16;
        Pack8 p0, p1;
        p0.q = *(const u32x4*)(bg);
        p1.q = *(const u32x4*)(bg + 8);
        if (ko + 32 < Kd) {                    // prefetch next K-slab
            __builtin_prefetch(ag + 32, 0, 1);
            __builtin_prefetch(bg + (size_t)32 * N, 0, 1);
        }
        __syncthreads();   // previous tile fully consumed by all waves
        // CDNA5 async copy: 2 x 16B per lane, offset advances both LDS + global
        asm volatile(
            "global_load_async_to_lds_b128 %0, %1, off\n\t"
            "global_load_async_to_lds_b128 %0, %1, off offset:16"
            :: "v"(aLds), "v"((unsigned long long)(uintptr_t)ag)
            : "memory");
        #pragma unroll
        for (int j = 0; j < 8; ++j) Bs[(bSeg * 16 + j)     * 40 + bK] = p0.h[j];
        #pragma unroll
        for (int j = 0; j < 8; ++j) Bs[(bSeg * 16 + 8 + j) * 40 + bK] = p1.h[j];
        asm volatile("s_wait_asynccnt 0" ::: "memory");
        __syncthreads();

        FragU fa[2], fb[4];
        #pragma unroll
        for (int rt = 0; rt < 2; ++rt) {        // A frag: k-groups {0..7,16..23}/{8..15,24..31}
            int base = (waveRow * 32 + rt * 16 + l15) * 40 + lHalf * 8;
            fa[rt].q[0] = *(const u32x4*)&As[base];
            fa[rt].q[1] = *(const u32x4*)&As[base + 16];
        }
        #pragma unroll
        for (int ct = 0; ct < 4; ++ct) {        // B frag: k-groups {0..7,16..23}/{8..15,24..31}
            int base = (waveCol * 64 + ct * 16 + l15) * 40 + lHalf * 8;
            fb[ct].q[0] = *(const u32x4*)&Bs[base];
            fb[ct].q[1] = *(const u32x4*)&Bs[base + 16];
        }
        #pragma unroll
        for (int rt = 0; rt < 2; ++rt)
            #pragma unroll
            for (int ct = 0; ct < 4; ++ct)
                acc[rt][ct] = __builtin_amdgcn_wmma_f32_16x16x32_f16(
                    false, fa[rt].v, false, fb[ct].v, (short)0, acc[rt][ct], false, false);
    }

    const int baseRow = rowBase + waveRow * 32;
    const int baseCol = colBase + waveCol * 64;

    if (MODE == 2) {
        #pragma unroll
        for (int rt = 0; rt < 2; ++rt) {
            #pragma unroll
            for (int r = 0; r < 8; ++r) {
                int gRow = baseRow + rt * 16 + 8 * lHalf + r;
                float s = 0.f;
                #pragma unroll
                for (int ct = 0; ct < 4; ++ct) {
                    int gCol = baseCol + ct * 16 + l15;
                    float v = acc[rt][ct][r] + bias[gCol];
                    v += P[(size_t)(gRow >> 5) * N + gCol];
                    v = v > 0.f ? v : 0.f;
                    s += v * vecW[gCol];
                }
                for (int off = 1; off < 16; off <<= 1) s += __shfl_xor(s, off, 32);
                if (l15 == 0) atomicAdd(accOut + gRow, s);
            }
        }
        return;
    }

    #pragma unroll
    for (int rt = 0; rt < 2; ++rt) {
        #pragma unroll
        for (int ct = 0; ct < 4; ++ct) {
            int gCol = baseCol + ct * 16 + l15;
            float bv = (MODE == 0) ? 0.f : bias[gCol];
            #pragma unroll
            for (int r = 0; r < 8; ++r) {
                int gRow = baseRow + rt * 16 + 8 * lHalf + r;
                size_t idx = (size_t)gRow * N + gCol;
                float v = acc[rt][ct][r] + bv;
                if (MODE == 0) {
                    outF[idx] = v;
                } else if (MODE == 1) {
                    if (RELU) v = v > 0.f ? v : 0.f;
                    outH[idx] = (_Float16)v;
                } else if (MODE == 3) {
                    int pair = gRow >> 1, tok = gRow & 1;
                    float res = tok ? resCand[(size_t)pair * N + gCol]
                                    : resMention[(size_t)(pair >> 5) * N + gCol];
                    outF[idx] = v + res;
                } else { // MODE 4
                    outF[idx] = v + resFull[idx];
                }
            }
        }
    }
}

// ---------------------------------------------------------------------------
__device__ inline float block_reduce_sum(float v, float* red) {
    int tid = threadIdx.x;
    red[tid] = v; __syncthreads();
    for (int s = 128; s > 0; s >>= 1) {
        if (tid < s) red[tid] += red[tid + s];
        __syncthreads();
    }
    float r = red[0]; __syncthreads();
    return r;
}

// mention/ctx span averages (direct row sums, spans <= 29 rows)
__global__ __launch_bounds__(256) void mention_ctx_kernel(
    const float* __restrict__ txt, const int* __restrict__ starts,
    const int* __restrict__ lens, float* __restrict__ mF, float* __restrict__ cF,
    _Float16* __restrict__ mH, _Float16* __restrict__ cH)
{
    int m = blockIdx.x, tid = threadIdx.x;
    int s = starts[m], L = lens[m], e = s + L;
    int cs = s - 10 < 0 ? 0 : s - 10;
    int ce = e + 10 > S_ - 1 ? S_ - 1 : e + 10;
    float invM = 1.f / (float)(L + 1), invC = 1.f / (float)(ce - cs);
    #pragma unroll
    for (int j = 0; j < 3; ++j) {
        int d = tid + j * 256;
        float am = 0.f, ac = 0.f;
        for (int r = s; r <= e; ++r)  am += txt[(size_t)r * D_ + d];
        for (int r = cs; r < ce; ++r) ac += txt[(size_t)r * D_ + d];
        am *= invM; ac *= invC;
        size_t o = (size_t)m * D_ + d;
        mF[o] = am; mH[o] = (_Float16)am;
        cF[o] = ac; cH[o] = (_Float16)ac;
    }
}

__global__ void cvt_f32_f16(const float* __restrict__ in, _Float16* __restrict__ out, int n) {
    int i = blockIdx.x * 256 + threadIdx.x;
    if (i < n) out[i] = (_Float16)in[i];
}

__global__ void zero_f32(float* __restrict__ p, int n) {
    int i = blockIdx.x * 256 + threadIdx.x;
    if (i < n) p[i] = 0.f;
}

// row-mean of uni_w2 (mean over output dim is linear -> GEMV vector), + mean(uni_b2)
__global__ void w2bar_kernel(const float* __restrict__ w2, const float* __restrict__ b2,
                             float* __restrict__ outv) {
    int j = blockIdx.x * 256 + threadIdx.x;
    if (j < D_) {
        float s = 0.f;
        for (int o = 0; o < D_; ++o) s += w2[(size_t)j * D_ + o];
        outv[j] = s * (1.f / (float)D_);
    }
    if (j == 0) {
        float s = 0.f;
        for (int o = 0; o < D_; ++o) s += b2[o];
        outv[D_] = s * (1.f / (float)D_);
    }
}

// 2-token, 8-head attention per (m,k) pair
__global__ __launch_bounds__(256) void attn_kernel(
    const _Float16* __restrict__ mq, const _Float16* __restrict__ mkk,
    const _Float16* __restrict__ mv, const _Float16* __restrict__ cq,
    const _Float16* __restrict__ ck, const _Float16* __restrict__ cv,
    _Float16* __restrict__ o)
{
    int pair = blockIdx.x, tid = threadIdx.x;
    int m = pair >> 5;
    __shared__ float sdot[4][8];
    __shared__ float aw[4][8];
    if (tid < 32) ((float*)sdot)[tid] = 0.f;
    __syncthreads();
    float q0[3], k0[3], v0[3], q1[3], k1[3], v1[3];
    #pragma unroll
    for (int j = 0; j < 3; ++j) {
        int d = tid + j * 256;
        size_t mo = (size_t)m * D_ + d, co = (size_t)pair * D_ + d;
        q0[j] = (float)mq[mo]; k0[j] = (float)mkk[mo]; v0[j] = (float)mv[mo];
        q1[j] = (float)cq[co]; k1[j] = (float)ck[co];  v1[j] = (float)cv[co];
        int h = d / 96;
        atomicAdd(&sdot[0][h], q0[j] * k0[j]);
        atomicAdd(&sdot[1][h], q0[j] * k1[j]);
        atomicAdd(&sdot[2][h], q1[j] * k0[j]);
        atomicAdd(&sdot[3][h], q1[j] * k1[j]);
    }
    __syncthreads();
    if (tid < 8) {
        const float sc = 0.10206207262f; // 1/sqrt(96)
        float s00 = sdot[0][tid] * sc, s01 = sdot[1][tid] * sc;
        float s10 = sdot[2][tid] * sc, s11 = sdot[3][tid] * sc;
        float m0 = fmaxf(s00, s01), e0 = __expf(s00 - m0), e1 = __expf(s01 - m0);
        float z0 = e0 + e1;
        aw[0][tid] = e0 / z0; aw[1][tid] = e1 / z0;
        float m1 = fmaxf(s10, s11), f0 = __expf(s10 - m1), f1 = __expf(s11 - m1);
        float z1 = f0 + f1;
        aw[2][tid] = f0 / z1; aw[3][tid] = f1 / z1;
    }
    __syncthreads();
    #pragma unroll
    for (int j = 0; j < 3; ++j) {
        int d = tid + j * 256, h = d / 96;
        float o0 = aw[0][h] * v0[j] + aw[1][h] * v1[j];
        float o1 = aw[2][h] * v0[j] + aw[3][h] * v1[j];
        size_t base = (size_t)pair * 2 * D_;
        o[base + d]      = (_Float16)o0;
        o[base + D_ + d] = (_Float16)o1;
    }
}

// LayerNorm in-place on f32 rows; also emits f16 copy
__global__ __launch_bounds__(256) void ln_kernel(
    float* __restrict__ x, _Float16* __restrict__ xh,
    const float* __restrict__ g, const float* __restrict__ b)
{
    __shared__ float red[256];
    size_t row = blockIdx.x;
    int tid = threadIdx.x;
    float v[3]; float s = 0.f, ss = 0.f;
    #pragma unroll
    for (int j = 0; j < 3; ++j) {
        v[j] = x[row * D_ + tid + j * 256];
        s += v[j]; ss += v[j] * v[j];
    }
    s  = block_reduce_sum(s, red);
    ss = block_reduce_sum(ss, red);
    float mu = s * (1.f / (float)D_);
    float var = ss * (1.f / (float)D_) - mu * mu;
    float rs = rsqrtf(var + 1e-5f);
    #pragma unroll
    for (int j = 0; j < 3; ++j) {
        int d = tid + j * 256;
        float y = (v[j] - mu) * rs * g[d] + b[d];
        x[row * D_ + d] = y;
        xh[row * D_ + d] = (_Float16)y;
    }
}

// final LayerNorm (ln2) on a token pair + cosine similarity -> atg
__global__ __launch_bounds__(256) void lncos_kernel(
    const float* __restrict__ gbuf, const float* __restrict__ g2,
    const float* __restrict__ b2, float* __restrict__ atg, int pairBase)
{
    __shared__ float red[256];
    int pl = blockIdx.x, tid = threadIdx.x;
    float e[2][3];
    for (int tok = 0; tok < 2; ++tok) {
        float v[3]; float s = 0.f, ss = 0.f;
        #pragma unroll
        for (int j = 0; j < 3; ++j) {
            v[j] = gbuf[((size_t)pl * 2 + tok) * D_ + tid + j * 256];
            s += v[j]; ss += v[j] * v[j];
        }
        s  = block_reduce_sum(s, red);
        ss = block_reduce_sum(ss, red);
        float mu = s * (1.f / (float)D_);
        float var = ss * (1.f / (float)D_) - mu * mu;
        float rs = rsqrtf(var + 1e-5f);
        #pragma unroll
        for (int j = 0; j < 3; ++j) {
            int d = tid + j * 256;
            e[tok][j] = (v[j] - mu) * rs * g2[d] + b2[d];
        }
    }
    float dot = 0.f, n0 = 0.f, n1 = 0.f;
    #pragma unroll
    for (int j = 0; j < 3; ++j) {
        dot += e[0][j] * e[1][j];
        n0  += e[0][j] * e[0][j];
        n1  += e[1][j] * e[1][j];
    }
    dot = block_reduce_sum(dot, red);
    n0  = block_reduce_sum(n0, red);
    n1  = block_reduce_sum(n1, red);
    if (tid == 0) {
        float a = fmaxf(sqrtf(n0), 1e-8f);
        float b = fmaxf(sqrtf(n1), 1e-8f);
        atg[pairBase + pl] = dot / (a * b);
    }
}

__global__ void finalize_kernel(const float* __restrict__ relikAcc,
                                const float* __restrict__ uniAcc,
                                const float* __restrict__ relik_b2,
                                const float* __restrict__ b2mean,
                                float* __restrict__ out) {
    int i = blockIdx.x * 256 + threadIdx.x;
    if (i < MK) {
        out[i] = relikAcc[i] + relik_b2[0];
        float u = uniAcc[i] + b2mean[0];
        out[2 * MK + i] = 1.f / (1.f + __expf(-u));
    }
}

// ---------------------------------------------------------------------------
extern "C" void kernel_launch(void* const* d_in, const int* in_sizes, int n_in,
                              void* d_out, int out_size, void* d_ws, size_t ws_size,
                              hipStream_t stream) {
    (void)in_sizes; (void)n_in; (void)out_size; (void)ws_size;
    const float* txt      = (const float*)d_in[0];
    const float* cand     = (const float*)d_in[1];
    const int*   starts   = (const int*)d_in[2];
    const int*   lens     = (const int*)d_in[3];
    const float* relik_w1 = (const float*)d_in[4];
    const float* relik_b1 = (const float*)d_in[5];
    const float* relik_w2 = (const float*)d_in[6];
    const float* relik_b2 = (const float*)d_in[7];
    const float* wq = (const float*)d_in[8];  const float* bq = (const float*)d_in[9];
    const float* wk = (const float*)d_in[10]; const float* bk = (const float*)d_in[11];
    const float* wv = (const float*)d_in[12]; const float* bv = (const float*)d_in[13];
    const float* wo = (const float*)d_in[14]; const float* bo = (const float*)d_in[15];
    const float* ln1_g = (const float*)d_in[16]; const float* ln1_b = (const float*)d_in[17];
    const float* ffn_w1 = (const float*)d_in[18]; const float* ffn_b1 = (const float*)d_in[19];
    const float* ffn_w2 = (const float*)d_in[20]; const float* ffn_b2 = (const float*)d_in[21];
    const float* ln2_g = (const float*)d_in[22]; const float* ln2_b = (const float*)d_in[23];
    const float* uni_w1 = (const float*)d_in[24]; const float* uni_b1 = (const float*)d_in[25];
    const float* uni_w2 = (const float*)d_in[26]; const float* uni_b2 = (const float*)d_in[27];
    float* out = (float*)d_out;

    char* wsp = (char*)d_ws;
    auto alloc = [&](size_t bytes) -> void* {
        void* p = (void*)wsp;
        wsp += (bytes + 255) & ~(size_t)255;
        return p;
    };
    float*    mentionF = (float*)alloc((size_t)M_ * D_ * 4);
    float*    ctxF     = (float*)alloc((size_t)M_ * D_ * 4);
    _Float16* mentionH = (_Float16*)alloc((size_t)M_ * D_ * 2);
    _Float16* ctxH     = (_Float16*)alloc((size_t)M_ * D_ * 2);
    _Float16* candH    = (_Float16*)alloc((size_t)MK * D_ * 2);
    _Float16* relikW1h = (_Float16*)alloc((size_t)2 * D_ * D_ * 2);
    _Float16* uniW1h   = (_Float16*)alloc((size_t)2 * D_ * D_ * 2);
    _Float16* wqh      = (_Float16*)alloc((size_t)D_ * D_ * 2);
    _Float16* wkh      = (_Float16*)alloc((size_t)D_ * D_ * 2);
    _Float16* wvh      = (_Float16*)alloc((size_t)D_ * D_ * 2);
    _Float16* woh      = (_Float16*)alloc((size_t)D_ * D_ * 2);
    _Float16* ffnW1h   = (_Float16*)alloc((size_t)D_ * FFN_H * 2);
    _Float16* ffnW2h   = (_Float16*)alloc((size_t)FFN_H * D_ * 2);
    float*    w2barF   = (float*)alloc(1024 * 4);
    float*    Pr       = (float*)alloc((size_t)M_ * D_ * 4);
    float*    Pu       = (float*)alloc((size_t)M_ * D_ * 4);
    float*    relikAcc = (float*)alloc((size_t)MK * 4);
    float*    uniAcc   = (float*)alloc((size_t)MK * 4);
    _Float16* mentQ    = (_Float16*)alloc((size_t)M_ * D_ * 2);
    _Float16* mentK    = (_Float16*)alloc((size_t)M_ * D_ * 2);
    _Float16* mentV    = (_Float16*)alloc((size_t)M_ * D_ * 2);
    _Float16* candQ    = (_Float16*)alloc((size_t)MK * D_ * 2);
    _Float16* candK    = (_Float16*)alloc((size_t)MK * D_ * 2);
    _Float16* candV    = (_Float16*)alloc((size_t)MK * D_ * 2);
    _Float16* oH       = (_Float16*)alloc((size_t)ROWS2 * D_ * 2);
    float*    x1F      = (float*)alloc((size_t)ROWS2 * D_ * 4);
    _Float16* x1H      = (_Float16*)alloc((size_t)ROWS2 * D_ * 2);
    _Float16* fH       = (_Float16*)alloc((size_t)CHUNK_ROWS * FFN_H * 2);
    float*    gF       = (float*)alloc((size_t)CHUNK_ROWS * D_ * 4);

    auto cvt = [&](const float* in, _Float16* o, int n) {
        cvt_f32_f16<<<(n + 255) / 256, 256, 0, stream>>>(in, o, n);
    };

    // 0) zero atomic accumulators
    zero_f32<<<(MK + 255) / 256, 256, 0, stream>>>(relikAcc, MK);
    zero_f32<<<(MK + 255) / 256, 256, 0, stream>>>(uniAcc, MK);

    // 1) mention / ctx span averages
    mention_ctx_kernel<<<M_, 256, 0, stream>>>(txt, starts, lens, mentionF, ctxF, mentionH, ctxH);

    // 2) f32 -> f16 conversions
    cvt(cand, candH, MK * D_);
    cvt(relik_w1, relikW1h, 2 * D_ * D_);
    cvt(uni_w1, uniW1h, 2 * D_ * D_);
    cvt(wq, wqh, D_ * D_);  cvt(wk, wkh, D_ * D_);
    cvt(wv, wvh, D_ * D_);  cvt(wo, woh, D_ * D_);
    cvt(ffn_w1, ffnW1h, D_ * FFN_H);
    cvt(ffn_w2, ffnW2h, FFN_H * D_);
    w2bar_kernel<<<3, 256, 0, stream>>>(uni_w2, uni_b2, w2barF);

    // 3) per-mention halves of the concat GEMMs: Pr = mention@W1a, Pu = ctx@U1a
    gemm_wmma_f16<0,0><<<dim3(D_/128, M_/128), 256, 0, stream>>>(
        mentionH, relikW1h, M_, D_, D_, nullptr, nullptr, nullptr, nullptr,
        Pr, nullptr, nullptr, nullptr, nullptr);
    gemm_wmma_f16<0,0><<<dim3(D_/128, M_/128), 256, 0, stream>>>(
        ctxH, uniW1h, M_, D_, D_, nullptr, nullptr, nullptr, nullptr,
        Pu, nullptr, nullptr, nullptr, nullptr);

    // 4) relik & unirel main GEMMs with fused relu+GEMV epilogue
    gemm_wmma_f16<2,0><<<dim3(D_/128, MK/128), 256, 0, stream>>>(
        candH, relikW1h + (size_t)D_ * D_, MK, D_, D_, relik_b1, Pr, relik_w2,
        relikAcc, nullptr, nullptr, nullptr, nullptr, nullptr);
    gemm_wmma_f16<2,0><<<dim3(D_/128, MK/128), 256, 0, stream>>>(
        candH, uniW1h + (size_t)D_ * D_, MK, D_, D_, uni_b1, Pu, w2barF,
        uniAcc, nullptr, nullptr, nullptr, nullptr, nullptr);

    // 5) QKV projections (mention shared across K, candidates full)
    const float* bqkv[3] = {bq, bk, bv};
    const _Float16* wqkv[3] = {wqh, wkh, wvh};
    _Float16* mouts[3] = {mentQ, mentK, mentV};
    _Float16* couts[3] = {candQ, candK, candV};
    for (int i = 0; i < 3; ++i) {
        gemm_wmma_f16<1,0><<<dim3(D_/128, M_/128), 256, 0, stream>>>(
            mentionH, wqkv[i], M_, D_, D_, bqkv[i], nullptr, nullptr, nullptr,
            nullptr, mouts[i], nullptr, nullptr, nullptr);
        gemm_wmma_f16<1,0><<<dim3(D_/128, MK/128), 256, 0, stream>>>(
            candH, wqkv[i], MK, D_, D_, bqkv[i], nullptr, nullptr, nullptr,
            nullptr, couts[i], nullptr, nullptr, nullptr);
    }

    // 6) 2-token attention
    attn_kernel<<<MK, 256, 0, stream>>>(mentQ, mentK, mentV, candQ, candK, candV, oH);

    // 7) output projection + residual, then LN1
    gemm_wmma_f16<3,0><<<dim3(D_/128, ROWS2/128), 256, 0, stream>>>(
        oH, woh, ROWS2, D_, D_, bo, nullptr, nullptr, nullptr,
        x1F, nullptr, mentionF, cand, nullptr);
    ln_kernel<<<ROWS2, 256, 0, stream>>>(x1F, x1H, ln1_g, ln1_b);

    // 8) FFN in row chunks + fused LN2 + cosine
    for (int c = 0; c < NCHUNK; ++c) {
        size_t rb = (size_t)c * CHUNK_ROWS;
        gemm_wmma_f16<1,1><<<dim3(FFN_H/128, CHUNK_ROWS/128), 256, 0, stream>>>(
            x1H + rb * D_, ffnW1h, CHUNK_ROWS, FFN_H, D_, ffn_b1, nullptr, nullptr,
            nullptr, nullptr, fH, nullptr, nullptr, nullptr);
        gemm_wmma_f16<4,0><<<dim3(D_/128, CHUNK_ROWS/128), 256, 0, stream>>>(
            fH, ffnW2h, CHUNK_ROWS, D_, FFN_H, ffn_b2, nullptr, nullptr,
            nullptr, gF, nullptr, nullptr, nullptr, x1F + rb * D_);
        lncos_kernel<<<CHUNK_ROWS / 2, 256, 0, stream>>>(gF, ln2_g, ln2_b,
                                                         out + MK, c * (CHUNK_ROWS / 2));
    }

    // 9) final scalar transforms -> out[0], out[2]
    finalize_kernel<<<(MK + 255) / 256, 256, 0, stream>>>(relikAcc, uniAcc,
                                                          relik_b2, w2barF + D_, out);
}
